// _SelectiveScanSigmoidLUT_30983894073635
// MI455X (gfx1250) — matrix operand order repaired and checked by reading, not compiled, not run on hardware
//
#include <hip/hip_runtime.h>

// ---------------------------------------------------------------------------
// lam = sigmoid(einsum('bkd,tk->btd', x, W) + b)  -> GEMM per batch + sigmoid
// ys  = scan over t of  s = lam*s + (1-lam)*x     -> chunked affine scan
// B=8, T=K=2048, D=1024
//
// GEMM: 128x128 block tile, 4 waves (wave32), 64x64 per wave = 4x4 WMMA tiles
//       -> 16 v_wmma per K-step vs 16 ds_load_b128 fragment loads.
//       Double-buffered LDS: one barrier per K-step, global loads overlap WMMA.
// ---------------------------------------------------------------------------

typedef _Float16 v16h __attribute__((ext_vector_type(16)));
typedef _Float16 v8h  __attribute__((ext_vector_type(8)));
typedef _Float16 v4h  __attribute__((ext_vector_type(4)));
typedef float    v8f  __attribute__((ext_vector_type(8)));

#define BATCH 8
#define MDIM 2048   // t (output rows)
#define NDIM 1024   // d (output cols)
#define KDIM 2048   // k (reduction)

#define TM 128
#define TN 128
#define TK 32
#define LDA 40      // padded halves per sA row (multiple of 8 -> 16B-aligned frags)
#define LDB 40      // padded halves per sB row

union H16 { v16h v; v8h h[2]; };

__device__ __forceinline__ float f4c(const float4& v, int i) {
  return i == 0 ? v.x : i == 1 ? v.y : i == 2 ? v.z : v.w;
}

__global__ __launch_bounds__(128) void gemm_sigmoid_kernel(
    const float* __restrict__ X,     // [B, K, D]
    const float* __restrict__ Wm,    // [T, K] row-major
    const float* __restrict__ bias,  // [T]
    float* __restrict__ lam)         // [B, T, D]  (== d_out, overwritten later)
{
  __shared__ __align__(16) _Float16 sA[2][TM * LDA];   // [m][k]
  __shared__ __align__(16) _Float16 sB[2][TN * LDB];   // [n][k] (transposed x tile)

  const int tid    = threadIdx.x;
  const int lane   = tid & 31;
  const int wave   = tid >> 5;     // 0..3
  const int wm     = wave & 1;     // 2 waves along M, 64 rows each
  const int wn     = wave >> 1;    // 2 waves along N, 64 cols each
  const int laneN  = lane & 15;
  const int laneHi = lane >> 4;

  const int n0 = blockIdx.x * TN;
  const int m0 = blockIdx.y * TM;
  const int bb = blockIdx.z;

  const float* Xb = X + (size_t)bb * KDIM * NDIM;

  // global->LDS work split (128 threads)
  const int arow = tid;               // W row within tile, loads k 0..31
  const int bk   = (tid >> 4) * 4;    // k offset 0..28  (4 rows of x)
  const int bn   = (tid & 15) * 8;    // n offset 0..120 (8 cols of x)

  v8f acc[4][4];
#pragma unroll
  for (int i = 0; i < 4; ++i)
#pragma unroll
    for (int j = 0; j < 4; ++j) acc[i][j] = {};

  float4 ra[8];   // 32 floats of W row segment
  float4 rx[8];   // 4 k-rows x 8 n of x

  auto load_globals = [&](int kk) {
    const float4* aw = (const float4*)(Wm + (size_t)(m0 + arow) * KDIM + kk);
#pragma unroll
    for (int i = 0; i < 8; ++i) ra[i] = aw[i];
    const float* xb = Xb + (size_t)(kk + bk) * NDIM + n0 + bn;
#pragma unroll
    for (int j = 0; j < 4; ++j) {
      rx[j * 2 + 0] = *(const float4*)(xb + j * NDIM);
      rx[j * 2 + 1] = *(const float4*)(xb + j * NDIM + 4);
    }
  };

  auto store_tiles = [&](int buf) {
    _Float16* dA = &sA[buf][arow * LDA];
#pragma unroll
    for (int i = 0; i < 4; ++i) {
      v8h h = {(_Float16)ra[2 * i].x,     (_Float16)ra[2 * i].y,
               (_Float16)ra[2 * i].z,     (_Float16)ra[2 * i].w,
               (_Float16)ra[2 * i + 1].x, (_Float16)ra[2 * i + 1].y,
               (_Float16)ra[2 * i + 1].z, (_Float16)ra[2 * i + 1].w};
      *(v8h*)(dA + i * 8) = h;              // ds_store_b128
    }
    _Float16* dB = &sB[buf][bn * LDB + bk];
#pragma unroll
    for (int i = 0; i < 8; ++i) {           // i = n offset within this thread
      const int hi = i >> 2, c = i & 3;
      v4h t = {(_Float16)f4c(rx[0 * 2 + hi], c), (_Float16)f4c(rx[1 * 2 + hi], c),
               (_Float16)f4c(rx[2 * 2 + hi], c), (_Float16)f4c(rx[3 * 2 + hi], c)};
      *(v4h*)(dB + i * LDB) = t;            // ds_store_b64 (4 k-contiguous halves)
    }
  };

  // prologue: stage tile 0
  load_globals(0);
  store_tiles(0);
  __syncthreads();

  for (int kk = 0; kk < KDIM; kk += TK) {
    const int cur = (kk >> 5) & 1;
    const bool more = (kk + TK) < KDIM;

    if (more) load_globals(kk + TK);        // overlaps with WMMA below
    if (kk + 2 * TK < KDIM) {
      __builtin_prefetch(Wm + (size_t)(m0 + arow) * KDIM + kk + 2 * TK, 0, 1);
      __builtin_prefetch(Xb + (size_t)(kk + 2 * TK + bk) * NDIM + n0 + bn, 0, 1);
    }

    // ---- fragments per documented gfx1250 16-bit WMMA layouts
    H16 af[4], bf[4];
#pragma unroll
    for (int mt = 0; mt < 4; ++mt) {
      // A 16x32: lane -> M=lane&15; halves 0..7 at K=laneHi*8, 8..15 at K=16+laneHi*8
      const _Float16* pa = &sA[cur][(wm * 64 + mt * 16 + laneN) * LDA + laneHi * 8];
      af[mt].h[0] = *(const v8h*)pa;
      af[mt].h[1] = *(const v8h*)(pa + 16);
    }
#pragma unroll
    for (int nt = 0; nt < 4; ++nt) {
      // B 32x16: lane -> N=lane&15; halves 0..7 at K=laneHi*8, 8..15 at K=16+laneHi*8
      const _Float16* pb = &sB[cur][(wn * 64 + nt * 16 + laneN) * LDB + laneHi * 8];
      bf[nt].h[0] = *(const v8h*)pb;
      bf[nt].h[1] = *(const v8h*)(pb + 16);
    }
#pragma unroll
    for (int mt = 0; mt < 4; ++mt)
#pragma unroll
      for (int nt = 0; nt < 4; ++nt)
        acc[mt][nt] = __builtin_amdgcn_wmma_f32_16x16x32_f16(
            false, af[mt].v, false, bf[nt].v, (short)0, acc[mt][nt], false, false);

    if (more) store_tiles(cur ^ 1);
    __syncthreads();
  }

  // ---- epilogue: bias + sigmoid, write lam
#pragma unroll
  for (int mt = 0; mt < 4; ++mt)
#pragma unroll
    for (int nt = 0; nt < 4; ++nt) {
      const int gN = n0 + wn * 64 + nt * 16 + laneN;
#pragma unroll
      for (int r = 0; r < 8; ++r) {
        const int gM = m0 + wm * 64 + mt * 16 + r + 8 * laneHi;   // C/D layout
        float z = acc[mt][nt][r] + bias[gM];
        float l = 1.0f / (1.0f + __expf(-z));
        lam[((size_t)bb * MDIM + gM) * NDIM + gN] = l;
      }
    }
}

// ---------------------------------------------------------------------------
// Chunked affine scan:  s_t = a_t*s_{t-1} + u_t,  u_t = (1-a_t)*x_t
// 8 chunks of 256 per column; 65536 threads in passes 1/3.
// ---------------------------------------------------------------------------
#define CL 256
#define NC 8

__global__ __launch_bounds__(256) void scan_pass1(
    const float* __restrict__ X, const float* __restrict__ lam,
    float* __restrict__ wsA, float* __restrict__ wsU)
{
  const int id = blockIdx.x * 256 + threadIdx.x;     // (b, chunk, d)
  const int d = id & (NDIM - 1);
  const int c = (id >> 10) & (NC - 1);
  const int b = id >> 13;
  const size_t base = ((size_t)b * MDIM + c * CL) * NDIM + d;
  const float* pl = lam + base;
  const float* px = X + base;
  float A = 1.0f, U = 0.0f;
#pragma unroll 4
  for (int t = 0; t < CL; ++t) {
    float a = pl[(size_t)t * NDIM];
    float u = (1.0f - a) * px[(size_t)t * NDIM];
    A = a * A;
    U = a * U + u;
  }
  wsA[id] = A;
  wsU[id] = U;
}

__global__ __launch_bounds__(256) void scan_pass2(
    const float* __restrict__ wsA, const float* __restrict__ wsU,
    float* __restrict__ wsS)
{
  const int id = blockIdx.x * 256 + threadIdx.x;     // (b, d), 8192 threads
  const int d = id & (NDIM - 1);
  const int b = id >> 10;
  float s = 0.0f;
#pragma unroll
  for (int c = 0; c < NC; ++c) {
    const int idx = (b * NC + c) * NDIM + d;
    wsS[idx] = s;                    // state entering chunk c
    s = wsA[idx] * s + wsU[idx];
  }
}

__global__ __launch_bounds__(256) void scan_pass3(
    const float* __restrict__ X, float* __restrict__ lam_io,
    const float* __restrict__ wsS)
{
  const int id = blockIdx.x * 256 + threadIdx.x;
  const int d = id & (NDIM - 1);
  const int c = (id >> 10) & (NC - 1);
  const int b = id >> 13;
  const size_t base = ((size_t)b * MDIM + c * CL) * NDIM + d;
  float* pl = lam_io + base;
  const float* px = X + base;
  float s = wsS[id];
#pragma unroll 4
  for (int t = 0; t < CL; ++t) {
    float a = pl[(size_t)t * NDIM];
    float u = (1.0f - a) * px[(size_t)t * NDIM];
    s = a * s + u;
    pl[(size_t)t * NDIM] = s;        // overwrite lam with ys, in place
  }
}

// ---------------------------------------------------------------------------
extern "C" void kernel_launch(void* const* d_in, const int* in_sizes, int n_in,
                              void* d_out, int out_size, void* d_ws, size_t ws_size,
                              hipStream_t stream) {
  const float* X    = (const float*)d_in[0];   // [8, 2048, 1024]
  const float* Wm   = (const float*)d_in[1];   // [2048, 2048]
  const float* bias = (const float*)d_in[2];   // [2048]
  float* out = (float*)d_out;                  // lam, then ys in place
  float* ws  = (float*)d_ws;

  float* wsA = ws;                 // 65536 floats
  float* wsU = ws + 65536;         // 65536 floats
  float* wsS = ws + 131072;        // 65536 floats

  dim3 gridG(NDIM / TN, MDIM / TM, BATCH);     // 8 x 16 x 8 = 1024 blocks
  gemm_sigmoid_kernel<<<gridG, 128, 0, stream>>>(X, Wm, bias, out);

  scan_pass1<<<(BATCH * NC * NDIM) / 256, 256, 0, stream>>>(X, out, wsA, wsU);
  scan_pass2<<<(BATCH * NDIM) / 256, 256, 0, stream>>>(wsA, wsU, wsS);
  scan_pass3<<<(BATCH * NC * NDIM) / 256, 256, 0, stream>>>(X, out, wsS);
}
